// UV_Aggregator_35072702939233
// MI455X (gfx1250) — matrix operand order repaired and checked by reading, not compiled, not run on hardware
//
#include <hip/hip_runtime.h>
#include <hip/hip_bf16.h>

// ---------------------------------------------------------------------------
// UV_Aggregator for MI455X (gfx1250): bf16 WMMA MLP chain + masked softmax.
// One workgroup per batch element; 128 threads (4 wave32).
// Weights pre-swizzled in LDS into WMMA B-fragment order (2 x b128 per frag).
// att3 score computed with WMMA (column-replicated weight vector).
// ---------------------------------------------------------------------------

typedef __attribute__((ext_vector_type(16))) __bf16 v16bf;
typedef __attribute__((ext_vector_type(8)))  __bf16 v8bf;
typedef __attribute__((ext_vector_type(2)))  __bf16 v2bf;
typedef __attribute__((ext_vector_type(8)))  float  v8f;

#define Bsz   4096
#define Nn    200
#define Dd    64
#define NTILE 13      // ceil(200/16)
#define NWAVE 4
#define ROWSP 208     // padded rows (13*16)
#define LDX   136     // bf16 stride of staged [16 x 128] input tile (16B-aligned rows)
#define LDH   72      // bf16 stride of per-wave [16 x 64] activation buffers
#define LDO   66      // f32 stride of o_history [208 x 64]

// ---- float LDS region (element counts) ----
#define F_O      (ROWSP * LDO)          // 13728
#define F_SCORES (ROWSP)                // 208
#define F_RED    (128)
#define F_BIAS   (64 * 4 + 1)           // b_r1,b_r2,b_a1,b_a2,b_a3
#define F_PAD    (3)                    // align bf16 region to 16B
#define F_TOT    (F_O + F_SCORES + F_RED + F_BIAS + F_PAD)   // 14324
// ---- bf16 LDS region (element counts) ----
// packed weight planes: [KT*4 frags][32 lanes][8 bf16] each plane
#define H_W1P  (4 * 4 * 32 * 8)         // 4096 per plane
#define H_W2P  (2 * 4 * 32 * 8)         // 2048 per plane
#define H_A3P  (2 * 32 * 8)             // 512 per plane (2 k-frags, col-replicated)
#define H_UV   (64)
#define H_XT   (NWAVE * 16 * LDX)       // 8704
#define H_HA   (NWAVE * 16 * LDH)       // 4608
#define H_TOT  (2*H_W1P + 2*H_W2P + 2*H_W1P + 2*H_W2P + 2*H_A3P + H_UV + H_XT + 2*H_HA)

#define SMEM_BYTES (F_TOT * 4 + H_TOT * 2)

__device__ __forceinline__ v8f wmma_bf16(v16bf a, v16bf b, v8f c) {
  return __builtin_amdgcn_wmma_f32_16x16x32_bf16(
      false, a, false, b, (short)0, c, false, false);
}

__device__ __forceinline__ v16bf cat16(v8bf lo, v8bf hi) {
  return __builtin_shufflevector(lo, hi, 0, 1, 2, 3, 4, 5, 6, 7,
                                          8, 9, 10, 11, 12, 13, 14, 15);
}

// A fragment (16x32 bf16, M x K) from LDS row-major [16 x ldx].
__device__ __forceinline__ v16bf load_afrag(const __bf16* X, int ldx, int kt) {
  int lane = threadIdx.x & 31;
  int row  = lane & 15;
  int kb   = kt * 32 + ((lane >> 4) << 4);
  const __bf16* p = X + row * ldx + kb;
  v8bf lo = *(const v8bf*)p;          // ds_load_b128
  v8bf hi = *(const v8bf*)(p + 8);    // ds_load_b128
  return cat16(lo, hi);
}

// B fragment (32x16 bf16, K x N) from pre-swizzled planes.
__device__ __forceinline__ v16bf load_bfrag(const __bf16* P0, const __bf16* P1, int f) {
  int lane = threadIdx.x & 31;
  int off  = (f * 32 + lane) * 8;
  v8bf lo = *(const v8bf*)(P0 + off); // ds_load_b128
  v8bf hi = *(const v8bf*)(P1 + off); // ds_load_b128
  return cat16(lo, hi);
}

// Pack a K x 64 (k-major f32 global) weight into fragment-order bf16 planes.
__device__ __forceinline__ void pack_weight(const float* __restrict__ G, int KT,
                                            __bf16* __restrict__ P0,
                                            __bf16* __restrict__ P1, int tid) {
  int tot = KT * 4 * 32 * 8;          // elements per plane
  for (int i = tid; i < tot; i += 128) {
    int j    = i & 7;
    int lane = (i >> 3) & 31;
    int f    = i >> 8;                // kt*4 + nt
    int kt   = f >> 2, nt = f & 3;
    int col  = (lane & 15) + (nt << 4);
    int kb   = (kt << 5) + ((lane >> 4) << 4);
    P0[i] = (__bf16)G[(kb + j) * 64 + col];
    P1[i] = (__bf16)G[(kb + 8 + j) * 64 + col];
  }
}

// Packed bf16 ReLU store of a row pair (cvt_pk + pk_max_num_bf16).
__device__ __forceinline__ void store_relu_pair(__bf16* H, int i0, int i1,
                                                float x0, float x1) {
  v2bf p;
  p[0] = (__bf16)x0;
  p[1] = (__bf16)x1;
  v2bf z = {(__bf16)0.f, (__bf16)0.f};
  p = __builtin_elementwise_max(p, z);
  H[i0] = p[0];
  H[i1] = p[1];
}

__global__ __launch_bounds__(128, 1)
void uv_agg_kernel(const int* __restrict__ nodes,
                   const int* __restrict__ hist_uv,
                   const int* __restrict__ hist_mask,
                   const int* __restrict__ hist_r,
                   const float* __restrict__ v2e,
                   const float* __restrict__ u2e,
                   const float* __restrict__ r2e,
                   const float* __restrict__ w1w, const float* __restrict__ w1b,
                   const float* __restrict__ w2w, const float* __restrict__ w2b,
                   const float* __restrict__ a1w, const float* __restrict__ a1b,
                   const float* __restrict__ a2w, const float* __restrict__ a2b,
                   const float* __restrict__ a3w, const float* __restrict__ a3b,
                   float* __restrict__ out) {
  extern __shared__ __align__(16) char smem_raw[];
  float*  sO      = (float*)smem_raw;             // [208 x LDO] o_history f32
  float*  sScores = sO + F_O;                     // [208] raw scores
  float*  sRed    = sScores + F_SCORES;           // [128]
  float*  sB1     = sRed + F_RED;                 // [64]
  float*  sB2     = sB1 + 64;
  float*  sBA1    = sB2 + 64;
  float*  sBA2    = sBA1 + 64;
  float*  sB3     = sBA2 + 64;                    // [1]
  __bf16* sW1P0   = (__bf16*)(smem_raw + F_TOT * 4);
  __bf16* sW1P1   = sW1P0 + H_W1P;
  __bf16* sW2P0   = sW1P1 + H_W1P;
  __bf16* sW2P1   = sW2P0 + H_W2P;
  __bf16* sA1P0   = sW2P1 + H_W2P;
  __bf16* sA1P1   = sA1P0 + H_W1P;
  __bf16* sA2P0   = sA1P1 + H_W1P;
  __bf16* sA2P1   = sA2P0 + H_W2P;
  __bf16* sA3P0   = sA2P1 + H_W2P;                // att3 col-replicated frags
  __bf16* sA3P1   = sA3P0 + H_A3P;
  __bf16* sUV     = sA3P1 + H_A3P;                // [64]
  __bf16* sXt     = sUV + H_UV;                   // [NWAVE][16 x LDX]
  __bf16* sHa     = sXt + H_XT;                   // [NWAVE][16 x LDH]
  __bf16* sHb     = sHa + H_HA;

  const int b    = blockIdx.x;
  const int tid  = threadIdx.x;
  const int lane = tid & 31;
  const int wid  = tid >> 5;
  const int node = nodes[b];

  // ---------------- Phase 0: swizzle weights / biases / uv_rep -----------
  pack_weight(w1w, 4, sW1P0, sW1P1, tid);
  pack_weight(a1w, 4, sA1P0, sA1P1, tid);
  pack_weight(w2w, 2, sW2P0, sW2P1, tid);
  pack_weight(a2w, 2, sA2P0, sA2P1, tid);
  // att3 as a column-replicated B operand (col index ignored)
  for (int i = tid; i < H_A3P; i += 128) {
    int j    = i & 7;
    int lane2 = (i >> 3) & 31;
    int f    = i >> 8;                // kt
    int kb   = (f << 5) + ((lane2 >> 4) << 4);
    sA3P0[i] = (__bf16)a3w[kb + j];
    sA3P1[i] = (__bf16)a3w[kb + 8 + j];
  }
  for (int i = tid; i < 64; i += 128) {
    sB1[i]  = w1b[i];
    sB2[i]  = w2b[i];
    sBA1[i] = a1b[i];
    sBA2[i] = a2b[i];
    sUV[i]  = (__bf16)u2e[(size_t)node * Dd + i];
  }
  if (tid == 0) sB3[0] = a3b[0];
  __syncthreads();

  // ---------------- Phase 1/2: per-wave 16-row tiles through MLP ---------
  __bf16* Xw  = sXt + wid * 16 * LDX;
  __bf16* Haw = sHa + wid * 16 * LDH;
  __bf16* Hbw = sHb + wid * 16 * LDH;

  const int colb  = lane & 15;
  const int rbase = (lane >> 4) << 3;
  const v8f zero = {0.f, 0.f, 0.f, 0.f, 0.f, 0.f, 0.f, 0.f};
  const float b3v = sB3[0];

  // hoist per-lane bias values (col depends only on lane)
  float bc1[4], bc2[4], bca1[4], bca2[4];
#pragma unroll
  for (int nt = 0; nt < 4; ++nt) {
    int col = colb + nt * 16;
    bc1[nt]  = sB1[col];
    bc2[nt]  = sB2[col];
    bca1[nt] = sBA1[col];
    bca2[nt] = sBA2[col];
  }

  for (int t = wid; t < NTILE; t += NWAVE) {
    // ---- gather: X = [ v2e[hist_uv] | r2e[hist_r] ] as bf16 (unmasked; ----
    // ---- masked rows are zeroed by softmax weight = 0 in phase 3)      ----
    {
      int r    = lane >> 1;           // row in tile
      int half = lane & 1;            // 0 -> e_uv, 1 -> e_r
      int n    = t * 16 + r;
      const float* src = v2e;         // padded rows: arbitrary (weight = 0)
      if (n < Nn) {
        int gi = half ? hist_r[b * Nn + n] : hist_uv[b * Nn + n];
        src = (half ? r2e : v2e) + (size_t)gi * Dd;
      }
      __bf16* dst = Xw + r * LDX + half * 64;
      const float4* s4 = (const float4*)src;
#pragma unroll
      for (int j = 0; j < 8; ++j) {
        float4 v0 = s4[2 * j];
        float4 v1 = s4[2 * j + 1];
        v8bf pk;
        pk[0] = (__bf16)v0.x; pk[1] = (__bf16)v0.y;
        pk[2] = (__bf16)v0.z; pk[3] = (__bf16)v0.w;
        pk[4] = (__bf16)v1.x; pk[5] = (__bf16)v1.y;
        pk[6] = (__bf16)v1.z; pk[7] = (__bf16)v1.w;
        *(v8bf*)(dst + 8 * j) = pk;   // ds_store_b128
      }
    }

    v8f acc[4];

    // ---- layer 1: relu(X[16x128] @ W1[128x64] + b1) -> Haw bf16 ----
    acc[0] = acc[1] = acc[2] = acc[3] = zero;
#pragma unroll
    for (int kt = 0; kt < 4; ++kt) {
      v16bf a = load_afrag(Xw, LDX, kt);
#pragma unroll
      for (int nt = 0; nt < 4; ++nt)
        acc[nt] = wmma_bf16(a, load_bfrag(sW1P0, sW1P1, kt * 4 + nt), acc[nt]);
    }
#pragma unroll
    for (int nt = 0; nt < 4; ++nt) {
      int col = colb + nt * 16;
#pragma unroll
      for (int q = 0; q < 8; q += 2)
        store_relu_pair(Haw, (rbase + q) * LDH + col, (rbase + q + 1) * LDH + col,
                        acc[nt][q] + bc1[nt], acc[nt][q + 1] + bc1[nt]);
    }

    // ---- layer 2: o = relu(H @ W2 + b2) -> sO f32 + Hbw bf16 ----
    acc[0] = acc[1] = acc[2] = acc[3] = zero;
#pragma unroll
    for (int kt = 0; kt < 2; ++kt) {
      v16bf a = load_afrag(Haw, LDH, kt);
#pragma unroll
      for (int nt = 0; nt < 4; ++nt)
        acc[nt] = wmma_bf16(a, load_bfrag(sW2P0, sW2P1, kt * 4 + nt), acc[nt]);
    }
    {
      int grow0 = t * 16 + rbase;
#pragma unroll
      for (int nt = 0; nt < 4; ++nt) {
        int col = colb + nt * 16;
#pragma unroll
        for (int q = 0; q < 8; q += 2) {
          float v0 = fmaxf(acc[nt][q] + bc2[nt], 0.f);
          float v1 = fmaxf(acc[nt][q + 1] + bc2[nt], 0.f);
          sO[(grow0 + q) * LDO + col]     = v0;
          sO[(grow0 + q + 1) * LDO + col] = v1;
          v2bf p;
          p[0] = (__bf16)v0;
          p[1] = (__bf16)v1;
          Hbw[(rbase + q) * LDH + col]     = p[0];
          Hbw[(rbase + q + 1) * LDH + col] = p[1];
        }
      }
    }

    // ---- att1: relu([o | uv_rep] @ A1 + ba1) -> Haw ----
    acc[0] = acc[1] = acc[2] = acc[3] = zero;
#pragma unroll
    for (int kt = 0; kt < 2; ++kt) {
      v16bf a = load_afrag(Hbw, LDH, kt);
#pragma unroll
      for (int nt = 0; nt < 4; ++nt)
        acc[nt] = wmma_bf16(a, load_bfrag(sA1P0, sA1P1, kt * 4 + nt), acc[nt]);
    }
#pragma unroll
    for (int kt = 2; kt < 4; ++kt) {
      v16bf a = load_afrag(sUV, 0, kt - 2);   // row-broadcast uv_rep
#pragma unroll
      for (int nt = 0; nt < 4; ++nt)
        acc[nt] = wmma_bf16(a, load_bfrag(sA1P0, sA1P1, kt * 4 + nt), acc[nt]);
    }
#pragma unroll
    for (int nt = 0; nt < 4; ++nt) {
      int col = colb + nt * 16;
#pragma unroll
      for (int q = 0; q < 8; q += 2)
        store_relu_pair(Haw, (rbase + q) * LDH + col, (rbase + q + 1) * LDH + col,
                        acc[nt][q] + bca1[nt], acc[nt][q + 1] + bca1[nt]);
    }

    // ---- att2: relu(H @ A2 + ba2) -> Hbw ----
    acc[0] = acc[1] = acc[2] = acc[3] = zero;
#pragma unroll
    for (int kt = 0; kt < 2; ++kt) {
      v16bf a = load_afrag(Haw, LDH, kt);
#pragma unroll
      for (int nt = 0; nt < 4; ++nt)
        acc[nt] = wmma_bf16(a, load_bfrag(sA2P0, sA2P1, kt * 4 + nt), acc[nt]);
    }
#pragma unroll
    for (int nt = 0; nt < 4; ++nt) {
      int col = colb + nt * 16;
#pragma unroll
      for (int q = 0; q < 8; q += 2)
        store_relu_pair(Hbw, (rbase + q) * LDH + col, (rbase + q + 1) * LDH + col,
                        acc[nt][q] + bca2[nt], acc[nt][q + 1] + bca2[nt]);
    }

    // ---- att3 via WMMA: B = att3_w replicated into all 16 columns ----
    {
      v8f sacc = zero;
#pragma unroll
      for (int kt = 0; kt < 2; ++kt) {
        v16bf a = load_afrag(Hbw, LDH, kt);
        sacc = wmma_bf16(a, load_bfrag(sA3P0, sA3P1, kt), sacc);
      }
      // every column of D holds the row scores; lanes 0 & 16 store rows
      if (colb == 0) {
        int base = t * 16 + rbase;
#pragma unroll
        for (int q = 0; q < 8; ++q)
          sScores[base + q] = sacc[q] + b3v;
      }
    }
  }

  __syncthreads();

  // ---------------- Phase 3: mask + softmax + weighted sum ---------------
  float lmax = -3.0e38f;
  for (int n = tid; n < ROWSP; n += 128) {
    bool valid = (n < Nn) && (hist_mask[b * Nn + n] > 0);
    float sc = valid ? sScores[n] : -1e9f;
    sScores[n] = sc;
    lmax = fmaxf(lmax, sc);
  }
  sRed[tid] = lmax;
  __syncthreads();
#pragma unroll
  for (int s = 64; s > 0; s >>= 1) {
    if (tid < s) sRed[tid] = fmaxf(sRed[tid], sRed[tid + s]);
    __syncthreads();
  }
  float mx = sRed[0];
  __syncthreads();

  float lsum = 0.f;
  for (int n = tid; n < ROWSP; n += 128) {
    float e = __expf(sScores[n] - mx);   // masked/padded -> 0
    sScores[n] = e;
    lsum += e;
  }
  __syncthreads();
  sRed[tid] = lsum;
  __syncthreads();
#pragma unroll
  for (int s = 64; s > 0; s >>= 1) {
    if (tid < s) sRed[tid] += sRed[tid + s];
    __syncthreads();
  }
  float inv = 1.0f / sRed[0];
  __syncthreads();

  int d = tid & 63;
  int part = tid >> 6;                   // 0 or 1
  float acc = 0.f;
  for (int n = part; n < Nn; n += 2)
    acc += sScores[n] * sO[n * LDO + d];
  sRed[tid] = acc;
  __syncthreads();
  if (tid < 64)
    out[(size_t)b * Dd + d] = (sRed[tid] + sRed[tid + 64]) * inv;
}

extern "C" void kernel_launch(void* const* d_in, const int* in_sizes, int n_in,
                              void* d_out, int out_size, void* d_ws, size_t ws_size,
                              hipStream_t stream) {
  const int*   nodes    = (const int*)d_in[0];
  const int*   hist_uv  = (const int*)d_in[1];
  const int*   hist_m   = (const int*)d_in[2];
  const int*   hist_r   = (const int*)d_in[3];
  const float* v2e      = (const float*)d_in[4];
  const float* u2e      = (const float*)d_in[5];
  const float* r2e      = (const float*)d_in[6];
  const float* w1w      = (const float*)d_in[7];
  const float* w1b      = (const float*)d_in[8];
  const float* w2w      = (const float*)d_in[9];
  const float* w2b      = (const float*)d_in[10];
  const float* a1w      = (const float*)d_in[11];
  const float* a1b      = (const float*)d_in[12];
  const float* a2w      = (const float*)d_in[13];
  const float* a2b      = (const float*)d_in[14];
  const float* a3w      = (const float*)d_in[15];
  const float* a3b      = (const float*)d_in[16];
  float* out = (float*)d_out;

  (void)hipFuncSetAttribute((const void*)uv_agg_kernel,
                            hipFuncAttributeMaxDynamicSharedMemorySize,
                            (int)SMEM_BYTES);

  uv_agg_kernel<<<Bsz, 128, SMEM_BYTES, stream>>>(
      nodes, hist_uv, hist_m, hist_r, v2e, u2e, r2e,
      w1w, w1b, w2w, w2b, a1w, a1b, a2w, a2b, a3w, a3b, out);
}
